// TaskAlignedAssigner_33200097198631
// MI455X (gfx1250) — matrix operand order repaired and checked by reading, not compiled, not run on hardware
//
#include <hip/hip_runtime.h>
#include <hip/hip_fp16.h>

// Problem constants (match setup_inputs in the reference)
#define B_ 32
#define A_ 8400
#define G_ 64
#define C_ 80
#define TOPK_ 13
#define TILE_A 128

typedef __attribute__((ext_vector_type(16))) _Float16 v16h;
typedef __attribute__((ext_vector_type(8)))  float    v8f;

__device__ __forceinline__ float sigmoid_f(float x) {
    return 1.0f / (1.0f + __expf(-x));
}

__device__ __forceinline__ float iou_f(float px1, float py1, float px2, float py2, float parea,
                                       float gx1, float gy1, float gx2, float gy2, float garea) {
    float ix1 = fmaxf(px1, gx1), iy1 = fmaxf(py1, gy1);
    float ix2 = fminf(px2, gx2), iy2 = fminf(py2, gy2);
    float iw = fmaxf(ix2 - ix1, 0.0f), ih = fmaxf(iy2 - iy1, 0.0f);
    float inter = iw * ih;
    float uni = parea + garea - inter + 1e-7f;
    return inter / uni;
}

// ---------------------------------------------------------------------------
// Workspace init: clear the per-anchor candidate bitmask (B*A * 2 u32)
// ---------------------------------------------------------------------------
__global__ void zero_u32_kernel(unsigned int* __restrict__ p, int n) {
    int i = blockIdx.x * blockDim.x + threadIdx.x;
    if (i < n) p[i] = 0u;
}

// ---------------------------------------------------------------------------
// Phase 1: for each (b, g) pick the top-13 anchors by metric and set bit g in
// cand_mask[b][a]. One 256-thread block per (b,g). Metrics kept in LDS so the
// 13 selection rounds can invalidate winners with dynamic indexing.
// pd_scores column gathers are served by L2 (whole tensor fits in 192MB L2).
// Scores are rounded through f16 to match the WMMA path of phase 2 exactly.
// ---------------------------------------------------------------------------
__global__ __launch_bounds__(256) void topk_cand_kernel(
    const float* __restrict__ pd_scores, const float* __restrict__ pd_bboxes,
    const int* __restrict__ gt_labels, const float* __restrict__ gt_bboxes,
    const unsigned char* __restrict__ mask_gt, unsigned int* __restrict__ cand_mask)
{
    const int b = blockIdx.x >> 6;
    const int g = blockIdx.x & 63;
    if (!mask_gt[b * G_ + g]) return;   // invalid gt: contributes no candidates

    const int cls = gt_labels[b * G_ + g];
    const float* gb = gt_bboxes + ((size_t)b * G_ + g) * 4;
    const float gcx = gb[0], gcy = gb[1], gw = gb[2], gh = gb[3];
    const float gx1 = gcx - gw * 0.5f, gy1 = gcy - gh * 0.5f;
    const float gx2 = gcx + gw * 0.5f, gy2 = gcy + gh * 0.5f;
    const float garea = (gx2 - gx1) * (gy2 - gy1);

    __shared__ float s_m[33 * 256];
    __shared__ float s_rv[256];
    __shared__ int   s_ri[256];

    const int tid = threadIdx.x;
    const float* pb_base = pd_bboxes + (size_t)b * A_ * 4;
    const float* ps_base = pd_scores + (size_t)b * A_ * C_;

    for (int i = 0; i < 33; i++) {
        const int a = tid + i * 256;
        float m = -3.0e38f;
        if (a < A_) {
            const float4 p = *(const float4*)(pb_base + (size_t)a * 4);
            const float px1 = p.x - p.z * 0.5f, py1 = p.y - p.w * 0.5f;
            const float px2 = p.x + p.z * 0.5f, py2 = p.y + p.w * 0.5f;
            const float parea = (px2 - px1) * (py2 - py1);
            const float iou = iou_f(px1, py1, px2, py2, parea, gx1, gy1, gx2, gy2, garea);
            const float sc = (float)(_Float16)ps_base[(size_t)a * C_ + cls]; // f16 round = WMMA path
            const float i2 = iou * iou;
            m = sigmoid_f(sc) * (i2 * i2 * i2);  // ALPHA=1, BETA=6
        }
        s_m[i * 256 + tid] = m;
    }
    __syncthreads();

    unsigned int* cm = cand_mask + ((size_t)b * A_) * 2;
    const unsigned int bitw = (unsigned)g >> 5;
    const unsigned int bit = 1u << (g & 31);

    for (int k = 0; k < TOPK_; k++) {
        // per-thread argmax over its 33 strided anchors (tie -> lowest index)
        float bv = -3.0e38f; int bi = 0x7fffffff;
        for (int i = 0; i < 33; i++) {
            const float v = s_m[i * 256 + tid];
            const int a = tid + i * 256;
            if (v > bv || (v == bv && a < bi)) { bv = v; bi = a; }
        }
        s_rv[tid] = bv; s_ri[tid] = bi;
        __syncthreads();
        for (int s = 128; s > 0; s >>= 1) {
            if (tid < s) {
                const float v2 = s_rv[tid + s]; const int i2 = s_ri[tid + s];
                if (v2 > s_rv[tid] || (v2 == s_rv[tid] && i2 < s_ri[tid])) {
                    s_rv[tid] = v2; s_ri[tid] = i2;
                }
            }
            __syncthreads();
        }
        const int win = s_ri[0];
        if (tid == 0) atomicOr(&cm[(size_t)win * 2 + bitw], bit);
        if ((win & 255) == tid) s_m[(win >> 8) * 256 + tid] = -3.0e38f; // remove winner
        __syncthreads();
    }
}

// ---------------------------------------------------------------------------
// Phase 2: per-anchor conflict resolution + output emission.
// One block = 128 anchors of one batch. Each of the 8 waves computes its
// 16(anchor) x 96(class,padded) x 64(gt) slice of the reference einsum
// ('bac,bgc->bag') on the matrix units: 4 gt-tiles x 3 K-slices of
// v_wmma_f32_16x16x32_f16 (B = one-hot of gt labels, exact in f16; the only
// rounding is f32->f16 of the logit, mirrored in phase 1).
// ---------------------------------------------------------------------------
__global__ __launch_bounds__(256) void assign_kernel(
    const float* __restrict__ pd_scores, const float* __restrict__ pd_bboxes,
    const int* __restrict__ gt_labels, const float* __restrict__ gt_bboxes,
    const unsigned char* __restrict__ mask_gt, const unsigned int* __restrict__ cand_mask,
    float* __restrict__ out_labels, float* __restrict__ out_bboxes,
    float* __restrict__ out_scores, float* __restrict__ out_fg)
{
    const int blocksPer = (A_ + TILE_A - 1) / TILE_A;   // 66
    const int b = blockIdx.x / blocksPer;
    const int base = (blockIdx.x % blocksPer) * TILE_A;
    const int tid = threadIdx.x;

    __shared__ float s_sc[TILE_A][G_ + 1];              // stride 65: bank-conflict free
    __shared__ float s_gx1[G_], s_gy1[G_], s_gx2[G_], s_gy2[G_], s_garea[G_];
    __shared__ float s_gcx[G_], s_gcy[G_], s_gw[G_], s_gh[G_];
    __shared__ int   s_lab[G_];
    __shared__ int   s_val[G_];
    __shared__ int   s_albl[TILE_A];
    __shared__ float s_aiou[TILE_A];
    __shared__ float s_afg[TILE_A];
    __shared__ float s_abb[TILE_A][4];

    // Stage A: gt data -> LDS
    if (tid < G_) {
        const float* gb = gt_bboxes + ((size_t)b * G_ + tid) * 4;
        const float cx = gb[0], cy = gb[1], w = gb[2], h = gb[3];
        s_gcx[tid] = cx; s_gcy[tid] = cy; s_gw[tid] = w; s_gh[tid] = h;
        const float x1 = cx - w * 0.5f, y1 = cy - h * 0.5f;
        const float x2 = cx + w * 0.5f, y2 = cy + h * 0.5f;
        s_gx1[tid] = x1; s_gy1[tid] = y1; s_gx2[tid] = x2; s_gy2[tid] = y2;
        s_garea[tid] = (x2 - x1) * (y2 - y1);
        s_lab[tid] = gt_labels[b * G_ + tid];
        s_val[tid] = mask_gt[b * G_ + tid] ? 1 : 0;
    }
    __syncthreads();

    // Stage B: score_gt = scores x onehot^T via WMMA. Wave w handles anchors
    // [base + 16w, base + 16w + 16). Layouts per CDNA5 ISA 7.12.2 (wave32):
    //   A (f16 16x32): lanes 0-15 row M=lane, halves 0..7 = K 0..7, 8..15 = K 16..23
    //                  lanes 16-31 same rows,  halves 0..7 = K 8..15, 8..15 = K 24..31
    //   B (f16 32x16): lanes 0-15 hold K 0..7,16..23 of column N=lane; lanes 16-31 K 8..15,24..31
    //   D (f32 16x16): lane N=lane&15, VGPR r -> M = r + 8*(lane>=16)
    const int wave = tid >> 5;
    const int lane = tid & 31;
    const int lane16 = lane & 15;
    const int hi = (lane >= 16) ? 1 : 0;
    {
        const int row = base + wave * 16 + lane16;
        const bool rok = row < A_;
        const float* srow = pd_scores + ((size_t)b * A_ + row) * C_;
        const int koff = hi ? 8 : 0;
        v16h Am[3];
        #pragma unroll
        for (int ks = 0; ks < 3; ks++) {
            #pragma unroll
            for (int t = 0; t < 8; t++) {
                const int k0 = ks * 32 + koff + t;
                const int k1 = ks * 32 + 16 + koff + t;
                Am[ks][t]     = (rok && k0 < C_) ? (_Float16)srow[k0] : (_Float16)0.0f;
                Am[ks][t + 8] = (rok && k1 < C_) ? (_Float16)srow[k1] : (_Float16)0.0f;
            }
        }
        const int kb = hi ? 16 : 0;
        #pragma unroll
        for (int gt = 0; gt < 4; gt++) {
            const int lab = s_lab[gt * 16 + lane16];
            v8f acc = {};
            #pragma unroll
            for (int ks = 0; ks < 3; ks++) {
                v16h Bm;
                #pragma unroll
                for (int e = 0; e < 16; e++)
                    Bm[e] = (lab == ks * 32 + (e < 8 ? (kb >> 1) + e : 16 + (kb >> 1) + e - 8)) ? (_Float16)1.0f : (_Float16)0.0f;
                acc = __builtin_amdgcn_wmma_f32_16x16x32_f16(
                    false, Am[ks], false, Bm, (short)0, acc, false, false);
            }
            const int gcol = gt * 16 + lane16;
            const int mbase = wave * 16 + (hi ? 8 : 0);
            #pragma unroll
            for (int r = 0; r < 8; r++)
                s_sc[mbase + r][gcol] = acc[r];
        }
    }
    __syncthreads();

    // Stage C: per-anchor single-pass resolution over g
    if (tid < TILE_A) {
        const int a = base + tid;
        if (a < A_) {
            const float4 p = *(const float4*)(pd_bboxes + ((size_t)b * A_ + a) * 4);
            const float px1 = p.x - p.z * 0.5f, py1 = p.y - p.w * 0.5f;
            const float px2 = p.x + p.z * 0.5f, py2 = p.y + p.w * 0.5f;
            const float parea = (px2 - px1) * (py2 - py1);
            const unsigned int mw0 = cand_mask[((size_t)b * A_ + a) * 2 + 0];
            const unsigned int mw1 = cand_mask[((size_t)b * A_ + a) * 2 + 1];
            float best = -3.0e38f;
            int lastmax = -1, firstg = -1;
            float iou_last = 0.0f, iou_first = 0.0f;
            for (int g = 0; g < G_; g++) {
                const float iou = iou_f(px1, py1, px2, py2, parea,
                                        s_gx1[g], s_gy1[g], s_gx2[g], s_gy2[g], s_garea[g]);
                const float sc = s_sc[tid][g];
                const float i2 = iou * iou;
                const float m = s_val[g] ? (sigmoid_f(sc) * (i2 * i2 * i2)) : -1.0f;
                const int cb = (g < 32) ? (int)((mw0 >> g) & 1u) : (int)((mw1 >> (g - 32)) & 1u);
                if (cb && firstg < 0) { firstg = g; iou_first = iou; }
                if (m > best)                { best = m; lastmax = cb ? g : -1; iou_last = iou; }
                else if (m == best && cb)    { lastmax = g; iou_last = iou; }
            }
            const bool fg = (mw0 | mw1) != 0u;
            const int asg = (lastmax >= 0) ? lastmax : firstg;
            if (fg && asg >= 0) {
                s_albl[tid] = s_lab[asg];
                s_aiou[tid] = (lastmax >= 0) ? iou_last : iou_first;
                s_afg[tid] = 1.0f;
                s_abb[tid][0] = s_gcx[asg]; s_abb[tid][1] = s_gcy[asg];
                s_abb[tid][2] = s_gw[asg];  s_abb[tid][3] = s_gh[asg];
            } else {
                s_albl[tid] = -1; s_aiou[tid] = 0.0f; s_afg[tid] = 0.0f;
                s_abb[tid][0] = s_abb[tid][1] = s_abb[tid][2] = s_abb[tid][3] = 0.0f;
            }
        }
    }
    __syncthreads();

    // Stage D: coalesced output emission (dominant HBM traffic)
    float* oL = out_labels + ((size_t)b * A_ + base) * C_;
    float* oS = out_scores + ((size_t)b * A_ + base) * C_;
    for (int idx = tid; idx < TILE_A * C_; idx += 256) {
        const int al = idx / C_, c = idx - al * C_;
        if (base + al < A_) {
            const float f = s_afg[al];
            const bool hit = (c == s_albl[al]);
            oL[(size_t)al * C_ + c] = hit ? f : 0.0f;
            oS[(size_t)al * C_ + c] = hit ? s_aiou[al] * f : 0.0f;
        }
    }
    if (tid < TILE_A && base + tid < A_) {
        const float4 bb = make_float4(s_abb[tid][0], s_abb[tid][1], s_abb[tid][2], s_abb[tid][3]);
        *(float4*)(out_bboxes + ((size_t)b * A_ + base + tid) * 4) = bb;
        out_fg[(size_t)b * A_ + base + tid] = s_afg[tid];
    }
}

// ---------------------------------------------------------------------------
extern "C" void kernel_launch(void* const* d_in, const int* in_sizes, int n_in,
                              void* d_out, int out_size, void* d_ws, size_t ws_size,
                              hipStream_t stream)
{
    (void)in_sizes; (void)n_in; (void)out_size; (void)ws_size;
    const float* pd_scores = (const float*)d_in[0];          // [B,A,C] f32
    const float* pd_bboxes = (const float*)d_in[1];          // [B,A,4] f32
    // d_in[2] = anc_points: unused by the reference computation
    const int*   gt_labels = (const int*)d_in[3];            // [B,G,1] i32
    const float* gt_bboxes = (const float*)d_in[4];          // [B,G,4] f32
    const unsigned char* mask_gt = (const unsigned char*)d_in[5]; // [B,G,1] bool (1 byte)

    float* out = (float*)d_out;                              // concat in return order
    float* out_labels = out;                                 // [B,A,C]
    float* out_bboxes = out_labels + (size_t)B_ * A_ * C_;   // [B,A,4]
    float* out_scores = out_bboxes + (size_t)B_ * A_ * 4;    // [B,A,C]
    float* out_fg     = out_scores + (size_t)B_ * A_ * C_;   // [B,A]

    unsigned int* cand_mask = (unsigned int*)d_ws;           // B*A*2 u32 = 2.15 MB
    const int nw = B_ * A_ * 2;
    zero_u32_kernel<<<(nw + 255) / 256, 256, 0, stream>>>(cand_mask, nw);

    topk_cand_kernel<<<B_ * G_, 256, 0, stream>>>(
        pd_scores, pd_bboxes, gt_labels, gt_bboxes, mask_gt, cand_mask);

    const int blocksPer = (A_ + TILE_A - 1) / TILE_A;        // 66
    assign_kernel<<<B_ * blocksPer, 256, 0, stream>>>(
        pd_scores, pd_bboxes, gt_labels, gt_bboxes, mask_gt, cand_mask,
        out_labels, out_bboxes, out_scores, out_fg);
}
